// OceanVQVAE_22325240004745
// MI455X (gfx1250) — matrix operand order repaired and checked by reading, not compiled, not run on hardware
//
#include <hip/hip_runtime.h>
#include <hip/hip_bf16.h>
#include <math.h>
#include <stdint.h>

// ---------------------------------------------------------------------------
// OceanVQVAE forward for MI455X (gfx1250, wave32).
//
// Roofline: ~7.6 GMAC total, ~250MB traffic -> ~11us at 23.3 TB/s; all
// intermediates fit in 192MB L2. Small-channel convs are direct fp32 kernels
// (latency/bandwidth bound; im2col+WMMA would add traffic at Cin=3..32).
// The VQ distance GEMM [60480x64]x[64x512] uses v_wmma_f32_16x16x32_f16:
// 32 rows per wave (4 WMMAs per 16-code column tile), codebook staged into
// LDS (64KB) once per workgroup via the Tensor Data Mover, B fragments read
// with ds_load_b128. argmin drops the per-row norm (constant across codes);
// vq_loss = (1+BETA)*mean((q-z)^2); q_st == q numerically.
// ---------------------------------------------------------------------------

typedef __attribute__((ext_vector_type(16))) _Float16 v16h;
typedef __attribute__((ext_vector_type(8)))  _Float16 v8h;
typedef __attribute__((ext_vector_type(8)))  float    v8f;
typedef unsigned int u32x4 __attribute__((ext_vector_type(4)));
typedef int          i32x4 __attribute__((ext_vector_type(4)));
typedef int          i32x8 __attribute__((ext_vector_type(8)));

#if defined(__has_builtin)
#if __has_builtin(__builtin_amdgcn_tensor_load_to_lds)
#define HAVE_TDM 1
#endif
#endif

#define BN_EPS 1e-5f

// ---- problem dims -----------------------------------------------------------
#define NB    16
#define H0    336
#define W0    720
#define LH    42
#define LW    90
#define NPIX  (NB*LH*LW)      // 60480 latent vectors
#define DEMB  64
#define KCODE 512

#define S_X   (NB*3*H0*W0)            // 11,612,160
#define S_H1  (NB*8*168*360)          // 7,741,440
#define S_H2  (NB*16*84*180)          // 3,870,720
#define S_H3  (NB*32*LH*LW)           // 1,935,360
#define S_LAT (NPIX*DEMB)             // 3,870,720

__device__ __forceinline__ float cleanf(float v) {
  return __builtin_isfinite(v) ? v : 0.0f;
}
__device__ __forceinline__ float lrelu(float v) {
  return v >= 0.0f ? v : 0.2f * v;
}

// ---- clean input ------------------------------------------------------------
__global__ void k_clean(const float* __restrict__ in, float* __restrict__ out, int n) {
  for (int i = blockIdx.x * blockDim.x + threadIdx.x; i < n; i += gridDim.x * blockDim.x)
    out[i] = cleanf(in[i]);
}

// ---- generic direct conv + BN + LeakyReLU (NCHW, weight OIHW) ---------------
template<int K, int S, int P>
__global__ void k_conv_bn_lrelu(const float* __restrict__ in, const float* __restrict__ w,
                                const float* __restrict__ bias,
                                const float* __restrict__ g, const float* __restrict__ be,
                                const float* __restrict__ m, const float* __restrict__ v,
                                float* __restrict__ out,
                                int Cin, int Cout, int Hin, int Win, int Hout, int Wout) {
  const int total = NB * Cout * Hout * Wout;
  for (int idx = blockIdx.x * blockDim.x + threadIdx.x; idx < total;
       idx += gridDim.x * blockDim.x) {
    int x = idx % Wout; int t = idx / Wout;
    int y = t % Hout;   t /= Hout;
    int co = t % Cout;  int n = t / Cout;
    float acc = bias[co];
    const float* wc = w + (size_t)co * Cin * K * K;
    for (int ci = 0; ci < Cin; ++ci) {
      const float* ip = in + (size_t)(n * Cin + ci) * Hin * Win;
      const float* wp = wc + ci * K * K;
#pragma unroll
      for (int ky = 0; ky < K; ++ky) {
        int iy = y * S - P + ky;
        if (iy < 0 || iy >= Hin) continue;
#pragma unroll
        for (int kx = 0; kx < K; ++kx) {
          int ix = x * S - P + kx;
          if (ix < 0 || ix >= Win) continue;
          acc = fmaf(ip[iy * Win + ix], wp[ky * K + kx], acc);
        }
      }
    }
    float inv = g[co] * rsqrtf(v[co] + BN_EPS);
    float val = acc * inv + (be[co] - m[co] * inv);
    out[idx] = lrelu(val);
  }
}

// ---- final encoder conv: 1x1 (32->64) + BN + tanh, emit NHWC latent (f32+f16)
__global__ void k_conv1x1_tanh_lat(const float* __restrict__ in, const float* __restrict__ w,
                                   const float* __restrict__ bias,
                                   const float* __restrict__ g, const float* __restrict__ be,
                                   const float* __restrict__ m, const float* __restrict__ v,
                                   float* __restrict__ lat, _Float16* __restrict__ lath) {
  const int total = NPIX * DEMB;
  for (int idx = blockIdx.x * blockDim.x + threadIdx.x; idx < total;
       idx += gridDim.x * blockDim.x) {
    int d = idx & 63;
    int nyx = idx >> 6;
    int n = nyx / (LH * LW);
    int rem = nyx % (LH * LW);
    float acc = bias[d];
    const float* ip = in + (size_t)n * 32 * LH * LW + rem;   // [16][32][42][90]
    const float* wp = w + d * 32;
#pragma unroll
    for (int ci = 0; ci < 32; ++ci)
      acc = fmaf(ip[(size_t)ci * LH * LW], wp[ci], acc);
    float inv = g[d] * rsqrtf(v[d] + BN_EPS);
    float val = tanhf(acc * inv + (be[d] - m[d] * inv));
    val = cleanf(val);
    lat[idx]  = val;
    lath[idx] = (_Float16)val;
  }
}

// ---- codebook prep: f16 copy + squared norms --------------------------------
__global__ void k_cb_prep(const float* __restrict__ cb, _Float16* __restrict__ cbh,
                          float* __restrict__ cn) {
  int k = blockIdx.x * blockDim.x + threadIdx.x;
  if (k >= KCODE) return;
  float s = 0.0f;
#pragma unroll 8
  for (int d = 0; d < DEMB; ++d) {
    float vv = cb[k * DEMB + d];
    s += vv * vv;
    cbh[k * DEMB + d] = (_Float16)vv;
  }
  cn[k] = s;
}

// ---- VQ argmin via WMMA f32_16x16x32_f16 ------------------------------------
// One wave per 32-row tile (two 16-row A fragment sets); 32 column tiles
// (512 codes) x 2 K-steps (K=64) -> 4 WMMAs per tile iteration. The 64KB f16
// codebook is staged to LDS once per workgroup by the Tensor Data Mover;
// B fragments then come from LDS (two 8-half chunks per lane per K-step).
// argmin(|z|^2+|c|^2-2z.c) == argmin(|c|^2-2z.c): row norms dropped.
__global__ void k_vq_argmin(const _Float16* __restrict__ lath,  // [NPIX][64]
                            const _Float16* __restrict__ cbh,   // [512][64]
                            const float* __restrict__ cn,       // [512]
                            int* __restrict__ inds, int nRows) {
  __shared__ _Float16 cbS[KCODE * DEMB];   // 65536 bytes

#if defined(HAVE_TDM)
  if (threadIdx.x < 32) {                  // wave 0 issues the DMA (EXEC ignored)
    unsigned ldsAddr = (unsigned)(uintptr_t)(void*)cbS;
    unsigned long long ga = (unsigned long long)(uintptr_t)cbh;
    // D# group0: count=1 | lds_addr | global_addr[56:0] | type=2 (bits 127:126)
    u32x4 g0 = { 1u, ldsAddr, (unsigned)ga,
                 (unsigned)((ga >> 32) & 0x1FFFFFFull) | (2u << 30) };
    // D# group1: data_size=2B (code 1) @bits17:16; tensor_dim0=64 @79:48;
    // tensor_dim1=512 @111:80; tile_dim0=64 @127:112; tile_dim1=512 @143:128;
    // tensor_dim0_stride=64 @207:160.
    i32x8 g1 = { 0x10000, 0x00400000, 0x02000000, 0x00400000, 0x200, 64, 0, 0 };
    i32x4 gz = { 0, 0, 0, 0 };
#if __clang_major__ >= 23
    i32x8 gz8 = { 0, 0, 0, 0, 0, 0, 0, 0 };
    __builtin_amdgcn_tensor_load_to_lds(g0, g1, gz, gz, gz8, 0);
#else
    __builtin_amdgcn_tensor_load_to_lds(g0, g1, gz, gz, 0);
#endif
    __builtin_amdgcn_s_wait_tensorcnt(0);
  }
  __syncthreads();
#else
  for (int i = threadIdx.x; i < KCODE * DEMB / 8; i += blockDim.x)
    ((v8h*)cbS)[i] = ((const v8h*)cbh)[i];
  __syncthreads();
#endif

  int wid  = (blockIdx.x * blockDim.x + threadIdx.x) >> 5;
  int lane = threadIdx.x & 31;
  int row0 = wid * 32;
  if (row0 >= nRows) return;             // uniform per wave (nRows % 32 == 0)
  int h = lane >> 4;                      // half id (0/1)
  int m = lane & 15;

  // A fragments (ISA 7.12.2, 16-bit A 16x32): lane holds row M=m,
  // dims {8h..8h+7} in a[0..7] and {16+8h..16+8h+7} in a[8..15].
  v16h a0, a1, a2, a3;
  {
    const _Float16* ar0 = lath + (size_t)(row0 + m) * DEMB;
    const _Float16* ar1 = lath + (size_t)(row0 + 16 + m) * DEMB;
    v8h c0 = *(const v8h*)(ar0 + 8 * h);
    v8h c1 = *(const v8h*)(ar0 + 16 + 8 * h);
    v8h c2 = *(const v8h*)(ar0 + 32 + 8 * h);
    v8h c3 = *(const v8h*)(ar0 + 48 + 8 * h);
    v8h d0 = *(const v8h*)(ar1 + 8 * h);
    v8h d1 = *(const v8h*)(ar1 + 16 + 8 * h);
    v8h d2 = *(const v8h*)(ar1 + 32 + 8 * h);
    v8h d3 = *(const v8h*)(ar1 + 48 + 8 * h);
#pragma unroll
    for (int i = 0; i < 8; ++i) {
      a0[i] = c0[i]; a0[i + 8] = c1[i];
      a1[i] = c2[i]; a1[i + 8] = c3[i];
      a2[i] = d0[i]; a2[i + 8] = d1[i];
      a3[i] = d2[i]; a3[i + 8] = d3[i];
    }
  }

  float bD0[8], bD1[8];
  int   bI0[8], bI1[8];
#pragma unroll
  for (int r = 0; r < 8; ++r) { bD0[r] = 3.4e38f; bI0[r] = 0; bD1[r] = 3.4e38f; bI1[r] = 0; }

#pragma unroll 2
  for (int t = 0; t < 32; ++t) {
    int code = t * 16 + m;               // B column owned by this lane
    const _Float16* brow = cbS + code * DEMB;   // LDS
    // B fragment (32x16): lane holds col N=m, K = {8h..8h+7} then {16+8h..16+8h+7}.
    v8h b0lo = *(const v8h*)(brow + 8 * h);
    v8h b0hi = *(const v8h*)(brow + 16 + 8 * h);
    v8h b1lo = *(const v8h*)(brow + 32 + 8 * h);
    v8h b1hi = *(const v8h*)(brow + 48 + 8 * h);
    v16h b0, b1;
#pragma unroll
    for (int i = 0; i < 8; ++i) {
      b0[i] = b0lo[i]; b0[i + 8] = b0hi[i];
      b1[i] = b1lo[i]; b1[i + 8] = b1hi[i];
    }
    v8f acc0 = {0.f, 0.f, 0.f, 0.f, 0.f, 0.f, 0.f, 0.f};
    v8f acc1 = {0.f, 0.f, 0.f, 0.f, 0.f, 0.f, 0.f, 0.f};
    acc0 = __builtin_amdgcn_wmma_f32_16x16x32_f16(false, a0, false, b0, (short)0, acc0, false, false);
    acc0 = __builtin_amdgcn_wmma_f32_16x16x32_f16(false, a1, false, b1, (short)0, acc0, false, false);
    acc1 = __builtin_amdgcn_wmma_f32_16x16x32_f16(false, a2, false, b0, (short)0, acc1, false, false);
    acc1 = __builtin_amdgcn_wmma_f32_16x16x32_f16(false, a3, false, b1, (short)0, acc1, false, false);
    // C layout: element r -> row (r + 8h), col m (same col whose B we loaded).
    float cnv = cn[code];
#pragma unroll
    for (int r = 0; r < 8; ++r) {
      float d0 = cnv - 2.0f * acc0[r];
      if (d0 < bD0[r]) { bD0[r] = d0; bI0[r] = code; }
      float d1 = cnv - 2.0f * acc1[r];
      if (d1 < bD1[r]) { bD1[r] = d1; bI1[r] = code; }
    }
  }

  // Reduce across the 16 lanes of each half (xor masks stay within the half).
#pragma unroll
  for (int off = 1; off < 16; off <<= 1) {
#pragma unroll
    for (int r = 0; r < 8; ++r) {
      float od0 = __shfl_xor(bD0[r], off, 32);
      int   oi0 = __shfl_xor(bI0[r], off, 32);
      if (od0 < bD0[r] || (od0 == bD0[r] && oi0 < bI0[r])) { bD0[r] = od0; bI0[r] = oi0; }
      float od1 = __shfl_xor(bD1[r], off, 32);
      int   oi1 = __shfl_xor(bI1[r], off, 32);
      if (od1 < bD1[r] || (od1 == bD1[r] && oi1 < bI1[r])) { bD1[r] = od1; bI1[r] = oi1; }
    }
  }
  if (m == 0) {
#pragma unroll
    for (int r = 0; r < 8; ++r) {
      inds[row0 + 8 * h + r]      = bI0[r];
      inds[row0 + 16 + 8 * h + r] = bI1[r];
    }
  }
}

// ---- gather q (NHWC->NCHW), accumulate ||q-z||^2, emit indices as float -----
__global__ void k_vq_gather(const int* __restrict__ inds, const float* __restrict__ cb,
                            const float* __restrict__ lat, float* __restrict__ qz,
                            float* __restrict__ lsum, float* __restrict__ indsF) {
  __shared__ float sdata[256];
  const int total = NPIX * DEMB;
  float local = 0.0f;
  for (int idx = blockIdx.x * blockDim.x + threadIdx.x; idx < total;
       idx += gridDim.x * blockDim.x) {
    int d = idx & 63;
    int nyx = idx >> 6;
    int ind = inds[nyx];
    float qv = cb[ind * DEMB + d];
    float diff = qv - lat[idx];
    local += diff * diff;
    int n = nyx / (LH * LW);
    int rem = nyx % (LH * LW);
    qz[((size_t)n * DEMB + d) * (LH * LW) + rem] = qv;   // NCHW for decoder
    if (d == 0) indsF[nyx] = (float)ind;
  }
  sdata[threadIdx.x] = local;
  __syncthreads();
  for (int s = 128; s > 0; s >>= 1) {
    if ((int)threadIdx.x < s) sdata[threadIdx.x] += sdata[threadIdx.x + s];
    __syncthreads();
  }
  if (threadIdx.x == 0) atomicAdd(lsum, sdata[0]);
}

__global__ void k_finalize_loss(const float* __restrict__ lsum, float* __restrict__ out) {
  float l = lsum[0] * 1.25f / (float)(NPIX * DEMB);   // (1+BETA)*mean
  out[0] = cleanf(l);
}

// ---- ConvTranspose2d k4 s2 p1 (+ optional BN+LReLU, else tanh+clean) --------
// torch weight layout [Cin][Cout][4][4]; out[o] += in[i]*w[k], o = 2i - 1 + k.
template<bool BN>
__global__ void k_convt(const float* __restrict__ in, const float* __restrict__ w,
                        const float* __restrict__ bias,
                        const float* __restrict__ g, const float* __restrict__ be,
                        const float* __restrict__ m, const float* __restrict__ v,
                        float* __restrict__ out, int Cin, int Cout, int Hin, int Win) {
  const int Hout = Hin * 2, Wout = Win * 2;
  const int total = NB * Cout * Hout * Wout;
  for (int idx = blockIdx.x * blockDim.x + threadIdx.x; idx < total;
       idx += gridDim.x * blockDim.x) {
    int ox = idx % Wout; int t = idx / Wout;
    int oy = t % Hout;   t /= Hout;
    int co = t % Cout;   int n = t / Cout;
    int ky0 = (oy + 1) & 1, kx0 = (ox + 1) & 1;
    int iy0 = (oy + 1 - ky0) >> 1, iy1 = iy0 - 1;   // taps ky0 and ky0+2
    int ix0 = (ox + 1 - kx0) >> 1, ix1 = ix0 - 1;
    bool vy0 = (iy0 >= 0 && iy0 < Hin), vy1 = (iy1 >= 0 && iy1 < Hin);
    bool vx0 = (ix0 >= 0 && ix0 < Win), vx1 = (ix1 >= 0 && ix1 < Win);
    float acc = bias[co];
    for (int ci = 0; ci < Cin; ++ci) {
      const float* ip = in + (size_t)(n * Cin + ci) * Hin * Win;
      const float* wp = w + (size_t)(ci * Cout + co) * 16;
      if (vy0) {
        if (vx0) acc = fmaf(ip[iy0 * Win + ix0], wp[ky0 * 4 + kx0], acc);
        if (vx1) acc = fmaf(ip[iy0 * Win + ix1], wp[ky0 * 4 + kx0 + 2], acc);
      }
      if (vy1) {
        if (vx0) acc = fmaf(ip[iy1 * Win + ix0], wp[(ky0 + 2) * 4 + kx0], acc);
        if (vx1) acc = fmaf(ip[iy1 * Win + ix1], wp[(ky0 + 2) * 4 + kx0 + 2], acc);
      }
    }
    if (BN) {
      float inv = g[co] * rsqrtf(v[co] + BN_EPS);
      out[idx] = lrelu(acc * inv + (be[co] - m[co] * inv));
    } else {
      out[idx] = cleanf(tanhf(acc));
    }
  }
}

// ---------------------------------------------------------------------------
static inline int nblk(long long total) { return (int)((total + 255) / 256); }

extern "C" void kernel_launch(void* const* d_in, const int* in_sizes, int n_in,
                              void* d_out, int out_size, void* d_ws, size_t ws_size,
                              hipStream_t stream) {
  (void)out_size; (void)ws_size;
  auto F = [&](int i) -> const float* { return (const float*)d_in[i]; };
  struct LP { const float *w, *b, *g, *be, *m, *v; };
  LP enc[6], dc[2], ct[3];
  const float* xin = nullptr;
  const float* cb  = nullptr;

  if (n_in >= 1 && in_sizes[0] == S_X) {
    // dict insertion order: x, enc[w,b,g,be,m,v]*6, dec_conv*2, dec_ct, codebook
    int i = 0; xin = F(i++);
    for (int L = 0; L < 6; ++L) { enc[L].w=F(i++); enc[L].b=F(i++); enc[L].g=F(i++); enc[L].be=F(i++); enc[L].m=F(i++); enc[L].v=F(i++); }
    for (int L = 0; L < 2; ++L) { dc[L].w=F(i++);  dc[L].b=F(i++);  dc[L].g=F(i++);  dc[L].be=F(i++);  dc[L].m=F(i++);  dc[L].v=F(i++); }
    for (int L = 0; L < 2; ++L) { ct[L].w=F(i++);  ct[L].b=F(i++);  ct[L].g=F(i++);  ct[L].be=F(i++);  ct[L].m=F(i++);  ct[L].v=F(i++); }
    ct[2].w = F(i++); ct[2].b = F(i++); ct[2].g = ct[2].be = ct[2].m = ct[2].v = nullptr;
    cb = F(i++);
  } else {
    // JAX pytree sorted order: codebook, dec_conv[b,be,g,m,v,w]*2, dec_ct, enc*6, x
    int i = 0; cb = F(i++);
    for (int L = 0; L < 2; ++L) { dc[L].b=F(i++);  dc[L].be=F(i++);  dc[L].g=F(i++);  dc[L].m=F(i++);  dc[L].v=F(i++);  dc[L].w=F(i++); }
    for (int L = 0; L < 2; ++L) { ct[L].b=F(i++);  ct[L].be=F(i++);  ct[L].g=F(i++);  ct[L].m=F(i++);  ct[L].v=F(i++);  ct[L].w=F(i++); }
    ct[2].b = F(i++); ct[2].w = F(i++); ct[2].g = ct[2].be = ct[2].m = ct[2].v = nullptr;
    for (int L = 0; L < 6; ++L) { enc[L].b=F(i++); enc[L].be=F(i++); enc[L].g=F(i++); enc[L].m=F(i++); enc[L].v=F(i++); enc[L].w=F(i++); }
    xin = F(i++);
  }

  // ---- workspace layout (float element offsets, buffers reused enc/dec) ----
  float* W = (float*)d_ws;
  float*     H1   = W;                      // 7,741,440  (also convT2 output)
  float*     H2   = W + 7741440;            // 3,870,720  (also convT1 output)
  float*     H3   = W + 11612160;           // 1,935,360  (also dec conv1 out)
  float*     H4   = W + 13547520;           // 1,935,360  (also dec conv2 out)
  float*     H5   = W + 15482880;           // 1,935,360
  float*     LAT  = W + 17418240;           // 3,870,720  NHWC f32 latent
  float*     QZ   = W + 21288960;           // 3,870,720  NCHW quantized
  _Float16*  LATH = (_Float16*)(W + 25159680);  // 3,870,720 halves
  _Float16*  CBH  = (_Float16*)(W + 27095040);  // 32,768 halves
  float*     CN   = W + 27111424;           // 512
  float*     LSUM = W + 27111936;           // 1
  int*       INDS = (int*)(W + 27111940);   // 60,480 ints

  float* out       = (float*)d_out;
  float* reconOut  = out;                   // 11,612,160
  float* xOut      = out + S_X;             // 11,612,160 (clean(x), also conv1 input)
  float* lossOut   = out + 2 * S_X;         // 1
  float* indsOut   = out + 2 * S_X + 1;     // 60,480

  // ---- encoder ----
  k_clean<<<nblk(S_X), 256, 0, stream>>>(xin, xOut, S_X);
  k_conv_bn_lrelu<4,2,1><<<nblk(S_H1), 256, 0, stream>>>(xOut, enc[0].w, enc[0].b,
      enc[0].g, enc[0].be, enc[0].m, enc[0].v, H1, 3, 8, 336, 720, 168, 360);
  k_conv_bn_lrelu<4,2,1><<<nblk(S_H2), 256, 0, stream>>>(H1, enc[1].w, enc[1].b,
      enc[1].g, enc[1].be, enc[1].m, enc[1].v, H2, 8, 16, 168, 360, 84, 180);
  k_conv_bn_lrelu<4,2,1><<<nblk(S_H3), 256, 0, stream>>>(H2, enc[2].w, enc[2].b,
      enc[2].g, enc[2].be, enc[2].m, enc[2].v, H3, 16, 32, 84, 180, 42, 90);
  k_conv_bn_lrelu<3,1,1><<<nblk(S_H3), 256, 0, stream>>>(H3, enc[3].w, enc[3].b,
      enc[3].g, enc[3].be, enc[3].m, enc[3].v, H4, 32, 32, 42, 90, 42, 90);
  k_conv_bn_lrelu<3,1,1><<<nblk(S_H3), 256, 0, stream>>>(H4, enc[4].w, enc[4].b,
      enc[4].g, enc[4].be, enc[4].m, enc[4].v, H5, 32, 32, 42, 90, 42, 90);
  k_conv1x1_tanh_lat<<<nblk(S_LAT), 256, 0, stream>>>(H5, enc[5].w, enc[5].b,
      enc[5].g, enc[5].be, enc[5].m, enc[5].v, LAT, LATH);

  // ---- vector quantization (TDM stage + WMMA) ----
  k_cb_prep<<<2, 256, 0, stream>>>(cb, CBH, CN);
  {
    int waves  = NPIX / 32;                 // 1890 row tiles (32 rows/wave), exact
    int blocks = (waves + 7) / 8;           // 8 waves per 256-thread block
    k_vq_argmin<<<blocks, 256, 0, stream>>>(LATH, CBH, CN, INDS, NPIX);
  }
  hipMemsetAsync(LSUM, 0, sizeof(float), stream);
  k_vq_gather<<<nblk(S_LAT), 256, 0, stream>>>(INDS, cb, LAT, QZ, LSUM, indsOut);
  k_finalize_loss<<<1, 1, 0, stream>>>(LSUM, lossOut);

  // ---- decoder ----
  k_conv_bn_lrelu<3,1,1><<<nblk(S_H3), 256, 0, stream>>>(QZ, dc[0].w, dc[0].b,
      dc[0].g, dc[0].be, dc[0].m, dc[0].v, H3, 64, 32, 42, 90, 42, 90);
  k_conv_bn_lrelu<3,1,1><<<nblk(S_H3), 256, 0, stream>>>(H3, dc[1].w, dc[1].b,
      dc[1].g, dc[1].be, dc[1].m, dc[1].v, H4, 32, 32, 42, 90, 42, 90);
  k_convt<true><<<nblk(S_H2), 256, 0, stream>>>(H4, ct[0].w, ct[0].b,
      ct[0].g, ct[0].be, ct[0].m, ct[0].v, H2, 32, 16, 42, 90);
  k_convt<true><<<nblk(S_H1), 256, 0, stream>>>(H2, ct[1].w, ct[1].b,
      ct[1].g, ct[1].be, ct[1].m, ct[1].v, H1, 16, 8, 84, 180);
  k_convt<false><<<nblk(S_X), 256, 0, stream>>>(H1, ct[2].w, ct[2].b,
      nullptr, nullptr, nullptr, nullptr, reconOut, 8, 3, 168, 360);
}
